// HierarchicalCircuitV3_27144193310739
// MI455X (gfx1250) — matrix operand order repaired and checked by reading, not compiled, not run on hardware
//
#include <hip/hip_runtime.h>
#include <hip/hip_bf16.h>
#include <math.h>
#include <stdint.h>

// ---------------------------------------------------------------------------
// HierarchicalCircuitV3 for MI455X (gfx1250, wave32, WMMA bf16 16x16x32).
//
//  * Step-invariant halves of all first-layer linears hoisted out of the
//    24-step loop (X fixed; only S_mean changes -> per-batch vector add).
//  * One-hot routing einsum -> global_atomic_add_f32 scatter.
//  * bf16 WMMA operands / fp32 accumulation; fp32 state + LayerNorm.
//  * k_msg and k_update stage weight k-slices into LDS with
//    TENSOR_LOAD_TO_LDS (TDM), double-buffered via s_wait_tensorcnt;
//    B fragments then come from LDS (ds_load_b128), next slice in flight.
// ---------------------------------------------------------------------------

#define BB     16
#define LTOT   1024      // 1023 edges + 1 query row
#define LEDGE  1023
#define DD     256
#define NSLOT  256
#define NSTEPS 24
#define HID    512

typedef __attribute__((ext_vector_type(16))) __bf16 bfrag;
typedef __attribute__((ext_vector_type(8)))  __bf16 v8bf;
typedef __attribute__((ext_vector_type(8)))  float  v8f;
typedef __attribute__((ext_vector_type(4)))  float  v4f;
typedef __attribute__((ext_vector_type(4)))  unsigned int v4u_t;
typedef __attribute__((ext_vector_type(8)))  int v8i_t;
typedef __attribute__((ext_vector_type(4)))  int v4i_t;

#if defined(__has_builtin)
#if __has_builtin(__builtin_amdgcn_tensor_load_to_lds) && \
    __has_builtin(__builtin_amdgcn_s_wait_tensorcnt)
#define HAVE_TDM 1
#endif
#endif
#ifndef HAVE_TDM
#define HAVE_TDM 0
#endif

__device__ __forceinline__ float geluf(float x) {
  return 0.5f * x * (1.0f + erff(x * 0.70710678118654752440f));
}

__device__ __forceinline__ __bf16 f2bf(float f) { return (__bf16)f; }

__device__ __forceinline__ v8f vzero8() {
  v8f z;
#pragma unroll
  for (int i = 0; i < 8; ++i) z[i] = 0.0f;
  return z;
}

__device__ __forceinline__ v8f wmma_bf16(bfrag a, bfrag b, v8f c) {
  return __builtin_amdgcn_wmma_f32_16x16x32_bf16(false, a, false, b,
                                                 (short)0, c, false, false);
}

// B fragment from row-major W[N][K] (torch Linear layout => computes A @ W^T).
__device__ __forceinline__ bfrag load_b_frag(const __bf16* W, int K, int n,
                                             int kb, int half) {
  const __bf16* p = W + (size_t)n * K + kb + (half ? 8 : 0);
  v8bf lo = *(const v8bf*)p;
  v8bf hi = *(const v8bf*)(p + 16);
  bfrag f;
#pragma unroll
  for (int i = 0; i < 8; ++i) { f[i] = lo[i]; f[8 + i] = hi[i]; }
  return f;
}

// A fragment from a row-major bf16 row (global or LDS).
__device__ __forceinline__ bfrag load_a_frag_bf16(const __bf16* row, int kb,
                                                  int half) {
  const __bf16* p0 = row + kb + (half ? 8 : 0);
  const __bf16* p1 = row + kb + 16 + (half ? 8 : 0);
  v8bf lo = *(const v8bf*)p0;
  v8bf hi = *(const v8bf*)p1;
  bfrag f;
#pragma unroll
  for (int i = 0; i < 8; ++i) { f[i] = lo[i]; f[8 + i] = hi[i]; }
  return f;
}

// A fragment built from an fp32 row with optional fused +sv[k] and GELU.
template <bool ADD, bool DO_GELU>
__device__ __forceinline__ bfrag make_a_frag_f32(const float* row,
                                                 const float* sv, int kb,
                                                 int half) {
  int k0 = kb + (half ? 8 : 0);
  int k1 = kb + 16 + (half ? 8 : 0);
  v4f a0 = *(const v4f*)(row + k0);
  v4f a1 = *(const v4f*)(row + k0 + 4);
  v4f b0 = *(const v4f*)(row + k1);
  v4f b1 = *(const v4f*)(row + k1 + 4);
  float t[16];
#pragma unroll
  for (int i = 0; i < 4; ++i) {
    t[i] = a0[i]; t[4 + i] = a1[i]; t[8 + i] = b0[i]; t[12 + i] = b1[i];
  }
  if (ADD) {
#pragma unroll
    for (int i = 0; i < 8; ++i) { t[i] += sv[k0 + i]; t[8 + i] += sv[k1 + i]; }
  }
  bfrag f;
#pragma unroll
  for (int i = 0; i < 16; ++i) {
    float x = t[i];
    if (DO_GELU) x = geluf(x);
    f[i] = f2bf(x);
  }
  return f;
}

#if HAVE_TDM
// TDM: DMA a 2D tile (d0 elements/row x d1 rows, 2-byte elements, row stride
// `stride0` elements) from global memory to LDS at byte offset `lds_off`.
// D# layout per CDNA5 ISA 8.3/8.4. Issued once per wave (EXEC ignored).
__device__ __forceinline__ void tdm_load_2d(unsigned lds_off, const void* g,
                                            unsigned d0, unsigned d1,
                                            unsigned stride0) {
  unsigned long long ga = (unsigned long long)(uintptr_t)g;
  v4u_t g0;
  g0[0] = 1u;                                   // count=1 (valid user D#)
  g0[1] = lds_off;                              // lds_addr (bytes)
  g0[2] = (unsigned)ga;                         // global_addr[31:0]
  g0[3] = (unsigned)((ga >> 32) & 0x01FFFFFFu)  // global_addr[56:32]
          | (2u << 30);                         // type = 2 ("image")
  v8i_t g1;
  g1[0] = (int)(1u << 16);                      // data_size=1 (2 bytes)
  g1[1] = (int)((d0 & 0xFFFFu) << 16);          // tensor_dim0 lo16
  g1[2] = (int)((d0 >> 16) | ((d1 & 0xFFFFu) << 16)); // dim0 hi | dim1 lo
  g1[3] = (int)((d1 >> 16) | ((d0 & 0xFFFFu) << 16)); // dim1 hi | tile_dim0
  g1[4] = (int)(d1 & 0xFFFFu);                  // tile_dim1 (tile_dim2=0)
  g1[5] = (int)stride0;                         // tensor_dim0_stride lo32
  g1[6] = 0;
  g1[7] = 0;
  v4i_t z4 = {0, 0, 0, 0};
#if __clang_major__ >= 23
  v8i_t z8 = {0, 0, 0, 0, 0, 0, 0, 0};
  __builtin_amdgcn_tensor_load_to_lds(g0, g1, z4, z4, z8, 0);
#else
  __builtin_amdgcn_tensor_load_to_lds(g0, g1, z4, z4, 0);
#endif
}
#endif

// ------------------------------ utility kernels ----------------------------

__global__ void k_cvt(__bf16* dst, const float* src, int n) {
  for (int i = blockIdx.x * blockDim.x + threadIdx.x; i < n;
       i += gridDim.x * blockDim.x)
    dst[i] = f2bf(src[i]);
}

__global__ void k_cvt_cols(__bf16* dst, const float* src, int rows, int scols,
                           int coff, int ncols) {
  int n = rows * ncols;
  for (int i = blockIdx.x * blockDim.x + threadIdx.x; i < n;
       i += gridDim.x * blockDim.x) {
    int r = i / ncols, c = i - r * ncols;
    dst[i] = f2bf(src[r * scols + coff + c]);
  }
}

__global__ void k_zero(float* p, int n) {
  for (int i = blockIdx.x * blockDim.x + threadIdx.x; i < n;
       i += gridDim.x * blockDim.x)
    p[i] = 0.0f;
}

__global__ void k_init_S(float* S, const float* S_init) {
  int n = BB * NSLOT * DD;
  for (int i = blockIdx.x * blockDim.x + threadIdx.x; i < n;
       i += gridDim.x * blockDim.x)
    S[i] = S_init[i & (NSLOT * DD - 1)];
}

// Query MLP -> X[b][1023][:] (bf16). One block per batch.
__global__ __launch_bounds__(256) void k_query(
    const int* qsrc, const int* qtgt, const float* node_embed,
    const float* w1, const float* b1, const float* w2, const float* b2,
    __bf16* X) {
  __shared__ float in[2 * DD];
  __shared__ float h[DD];
  int b = blockIdx.x, t = threadIdx.x;
  int qs = min(max(qsrc[b], 0), 49);
  int qt = min(max(qtgt[b], 0), 49);
  in[t] = node_embed[qs * DD + t];
  in[DD + t] = node_embed[qt * DD + t];
  __syncthreads();
  float a = b1[t];
  const float* wr = w1 + t * (2 * DD);
  for (int k = 0; k < 2 * DD; ++k) a += wr[k] * in[k];
  h[t] = geluf(a);
  __syncthreads();
  float o = b2[t];
  const float* w2r = w2 + t * DD;
  for (int k = 0; k < DD; ++k) o += w2r[k] * h[k];
  X[((size_t)b * LTOT + (LTOT - 1)) * DD + t] = f2bf(o);
}

// ------------------------------ edge MLP (once) ----------------------------
__global__ __launch_bounds__(128) void k_edge(
    const int* src, const int* rel, const int* tgt, const float* node_embed,
    const float* rel_embed, const __bf16* w1, const float* b1,
    const __bf16* w2, const float* b2, __bf16* X) {
  __shared__ __align__(16) __bf16 hl[4][16][DD];
  int wave = threadIdx.x >> 5, lane = threadIdx.x & 31;
  int half = lane >> 4, nl = lane & 15;
  int tile = blockIdx.x * 4 + wave;            // 0..1023
  int b = tile >> 6, l0 = (tile & 63) * 16;
  int m = nl;
  int lc = min(l0 + m, LEDGE - 1);
  int si = min(max(src[b * LEDGE + lc], 0), 49);
  int ri = rel[b * LEDGE + lc];
  int ti = min(max(tgt[b * LEDGE + lc], 0), 49);
  const float* ps = node_embed + (size_t)si * DD;
  const float* pr = rel_embed + (size_t)ri * DD;
  const float* pt = node_embed + (size_t)ti * DD;

  v8f acc[16];
#pragma unroll
  for (int t = 0; t < 16; ++t) acc[t] = vzero8();
  for (int ks = 0; ks < 24; ++ks) {            // K = 768
    int kb = ks * 32;
    int seg = kb >> 8;
    const float* rowp = (seg == 0) ? (ps + kb)
                       : (seg == 1) ? (pr + (kb - 256)) : (pt + (kb - 512));
    bfrag a = make_a_frag_f32<false, false>(rowp, nullptr, 0, half);
    bfrag bc = load_b_frag(w1, 768, nl, kb, half);
#pragma unroll
    for (int t = 0; t < 16; ++t) {
      bfrag bn = (t < 15) ? load_b_frag(w1, 768, (t + 1) * 16 + nl, kb, half)
                          : bc;
      acc[t] = wmma_bf16(a, bc, acc[t]);
      bc = bn;
    }
  }
#pragma unroll
  for (int t = 0; t < 16; ++t) {
    int n = t * 16 + nl;
    float bb = b1[n];
#pragma unroll
    for (int j = 0; j < 8; ++j)
      hl[wave][j + 8 * half][n] = f2bf(geluf(acc[t][j] + bb));
  }
  v8f acc2[16];
#pragma unroll
  for (int t = 0; t < 16; ++t) acc2[t] = vzero8();
  for (int ks = 0; ks < 8; ++ks) {             // K = 256, A from own LDS
    int kb = ks * 32;
    bfrag a = load_a_frag_bf16(&hl[wave][m][0], kb, half);
    bfrag bc = load_b_frag(w2, 256, nl, kb, half);
#pragma unroll
    for (int t = 0; t < 16; ++t) {
      bfrag bn = (t < 15) ? load_b_frag(w2, 256, (t + 1) * 16 + nl, kb, half)
                          : bc;
      acc2[t] = wmma_bf16(a, bc, acc2[t]);
      bc = bn;
    }
  }
#pragma unroll
  for (int t = 0; t < 16; ++t) {
    int n = t * 16 + nl;
    float bb = b2[n];
#pragma unroll
    for (int j = 0; j < 8; ++j) {
      int l = l0 + j + 8 * half;
      if (l < LEDGE)                           // row 1023 owned by k_query
        X[((size_t)b * LTOT + l) * DD + n] = f2bf(acc2[t][j] + bb);
    }
  }
}

// ---------------- generic C_f32 = A_bf16 @ W_bf16^T (+bias) ----------------
__global__ __launch_bounds__(128) void k_lin(const __bf16* A, const __bf16* W,
                                             const float* bias, float* C,
                                             int M, int N, int K) {
  int wid = blockIdx.x * (blockDim.x >> 5) + (threadIdx.x >> 5);
  int ntiles = N >> 4;
  int mt = wid / ntiles, nt = wid - mt * ntiles;
  if (mt * 16 >= M) return;
  int lane = threadIdx.x & 31, half = lane >> 4, nl = lane & 15;
  const __bf16* arow = A + (size_t)(mt * 16 + nl) * K;
  int n = nt * 16 + nl;
  v8f acc = vzero8();
  bfrag ac = load_a_frag_bf16(arow, 0, half);
  bfrag bc = load_b_frag(W, K, n, 0, half);
  for (int kb = 0; kb < K; kb += 32) {
    bfrag an = ac, bn = bc;
    if (kb + 32 < K) {
      an = load_a_frag_bf16(arow, kb + 32, half);
      bn = load_b_frag(W, K, n, kb + 32, half);
    }
    acc = wmma_bf16(ac, bc, acc);
    ac = an; bc = bn;
  }
  float bb = bias ? bias[n] : 0.0f;
#pragma unroll
  for (int j = 0; j < 8; ++j)
    C[(size_t)(mt * 16 + j + 8 * half) * N + n] = acc[j] + bb;
}

// ------------------- per-step S_mean projections (tiny) --------------------
__global__ __launch_bounds__(256) void k_sv(
    const float* S, const float* msg_w1, const float* gate_w1,
    const float* group_w, const float* slot_w, float* sv_msg, float* sv_gate,
    float* sv_grp, float* sv_slt) {
  __shared__ float sm[DD];
  int b = blockIdx.x, t = threadIdx.x;
  float a = 0.0f;
  for (int n = 0; n < NSLOT; ++n) a += S[((size_t)b * NSLOT + n) * DD + t];
  sm[t] = a * (1.0f / (float)NSLOT);
  __syncthreads();
  for (int j = t; j < HID; j += 256) {
    float d = 0.0f;
    const float* wr = msg_w1 + (size_t)j * (2 * DD) + DD;
    for (int k = 0; k < DD; ++k) d += sm[k] * wr[k];
    sv_msg[b * HID + j] = d;
  }
  {
    float d = 0.0f;
    const float* wr = gate_w1 + (size_t)t * (2 * DD) + DD;
    for (int k = 0; k < DD; ++k) d += sm[k] * wr[k];
    sv_gate[b * DD + t] = d;
  }
  if (t < 16) {
    float d = 0.0f;
    const float* wr = group_w + (size_t)t * (2 * DD) + DD;
    for (int k = 0; k < DD; ++k) d += sm[k] * wr[k];
    sv_grp[b * 16 + t] = d;
  } else if (t < 32) {
    float d = 0.0f;
    const float* wr = slot_w + (size_t)(t - 16) * (2 * DD) + DD;
    for (int k = 0; k < DD; ++k) d += sm[k] * wr[k];
    sv_slt[b * 16 + (t - 16)] = d;
  }
}

// -------------------- gate + hard routing (one wave/row) -------------------
__global__ __launch_bounds__(256) void k_route(
    const float* Hgate, const float* sv_gate, const float* gate_w2,
    const float* gate_b2, const float* Hgrp, const float* sv_grp,
    const float* Hslt, const float* sv_slt, float* gatev, int* nidx) {
  int wid = blockIdx.x * 8 + (threadIdx.x >> 5);   // 0..16383
  int lane = threadIdx.x & 31;
  int b = wid >> 10, l = wid & (LTOT - 1);
  const float* hr = Hgate + ((size_t)b * LTOT + l) * DD;
  const float* sg = sv_gate + b * DD;
  float part = 0.0f;
  v4f x0 = *(const v4f*)(hr + lane * 8);
  v4f x1 = *(const v4f*)(hr + lane * 8 + 4);
#pragma unroll
  for (int i = 0; i < 4; ++i) {
    int d = lane * 8 + i;
    part += geluf(x0[i] + sg[d]) * gate_w2[d];
    part += geluf(x1[i] + sg[d + 4]) * gate_w2[d + 4];
  }
#pragma unroll
  for (int off = 16; off >= 1; off >>= 1) part += __shfl_xor(part, off, 32);
  float v; int idx;
  if (lane < 16) {
    v = Hgrp[((size_t)b * LTOT + l) * 16 + lane] + sv_grp[b * 16 + lane];
    idx = lane;
  } else {
    int s = lane - 16;
    v = Hslt[((size_t)b * LTOT + l) * 16 + s] + sv_slt[b * 16 + s];
    idx = s;
  }
#pragma unroll
  for (int off = 8; off >= 1; off >>= 1) {
    float ov = __shfl_xor(v, off, 32);
    int oi = __shfl_xor(idx, off, 32);
    if (ov > v || (ov == v && oi < idx)) { v = ov; idx = oi; }
  }
  int slot = __shfl(idx, 16, 32);
  if (lane == 0) {
    gatev[b * LTOT + l] = 1.0f / (1.0f + expf(-(part + gate_b2[0])));
    nidx[b * LTOT + l] = idx * 16 + slot;
  }
}

// ------ msg lin2 (fused +sv, GELU, bf16) + gate scale + atomic scatter -----
__global__ __launch_bounds__(128) void k_msg(const float* Hmsg,
                                             const float* sv_msg,
                                             const __bf16* w2, const float* b2,
                                             const float* gatev,
                                             const int* nidx,
                                             float* incoming) {
  int wave = threadIdx.x >> 5;
  int lane = threadIdx.x & 31, half = lane >> 4, nl = lane & 15;
  int wid = blockIdx.x * 4 + wave;                 // 0..1023
  int b = wid >> 6, l0 = (wid & 63) * 16;
  const float* arow = Hmsg + ((size_t)b * LTOT + l0 + nl) * HID;
  const float* sv = sv_msg + b * HID;
  v8f acc[16];
#pragma unroll
  for (int t = 0; t < 16; ++t) acc[t] = vzero8();

#if HAVE_TDM
  __shared__ __align__(16) __bf16 wl[2][NSLOT * 32];   // 2 x 16 KB k-slices
  unsigned lo0 = (unsigned)(uintptr_t)(const void*)&wl[0][0];
  unsigned lo1 = (unsigned)(uintptr_t)(const void*)&wl[1][0];
  if (wave == 0)                                   // slice ks=0 -> buf0
    tdm_load_2d(lo0, w2, 32, NSLOT, HID);
  for (int ks = 0; ks < 16; ++ks) {                // K = 512
    int kb = ks * 32;
    if (wave == 0) {
      if (ks + 1 < 16) {                           // prefetch next slice
        tdm_load_2d(((ks + 1) & 1) ? lo1 : lo0, w2 + (size_t)(kb + 32), 32,
                    NSLOT, HID);
        __builtin_amdgcn_s_wait_tensorcnt(1);      // slice ks is resident
      } else {
        __builtin_amdgcn_s_wait_tensorcnt(0);
      }
    }
    __syncthreads();
    const __bf16* ws = &wl[ks & 1][0];
    bfrag a = make_a_frag_f32<true, true>(arow, sv, kb, half);
    bfrag bc = load_b_frag(ws, 32, nl, 0, half);
#pragma unroll
    for (int t = 0; t < 16; ++t) {
      bfrag bn = (t < 15) ? load_b_frag(ws, 32, (t + 1) * 16 + nl, 0, half)
                          : bc;
      acc[t] = wmma_bf16(a, bc, acc[t]);
      bc = bn;
    }
    __syncthreads();                               // slice consumed
  }
#else
  for (int ks = 0; ks < 16; ++ks) {
    int kb = ks * 32;
    bfrag a = make_a_frag_f32<true, true>(arow, sv, kb, half);
    bfrag bc = load_b_frag(w2, HID, nl, kb, half);
#pragma unroll
    for (int t = 0; t < 16; ++t) {
      bfrag bn = (t < 15) ? load_b_frag(w2, HID, (t + 1) * 16 + nl, kb, half)
                          : bc;
      acc[t] = wmma_bf16(a, bc, acc[t]);
      bc = bn;
    }
  }
#endif

  float g8[8]; int id8[8];
#pragma unroll
  for (int j = 0; j < 8; ++j) {
    int l = l0 + j + 8 * half;
    g8[j] = gatev[b * LTOT + l];
    id8[j] = nidx[b * LTOT + l];
  }
#pragma unroll
  for (int t = 0; t < 16; ++t) {
    int n = t * 16 + nl;
    float bb = b2[n];
#pragma unroll
    for (int j = 0; j < 8; ++j) {
      float val = (acc[t][j] + bb) * g8[j];
      atomicAdd(incoming + ((size_t)b * NSLOT + id8[j]) * DD + n, val);
    }
  }
}

// -------- slot update MLP (WMMA x2, LDS-staged hidden) + residual LN -------
// TDM-staged weight k-slices (double buffered): GEMM1 uses 2x32KB buffers,
// GEMM2 reuses the same region as 2x16KB buffers.
__global__ __launch_bounds__(128) void k_update(
    float* S, const float* incoming, const __bf16* w1, const float* b1,
    const __bf16* w2, const float* b2, const float* ln_g, const float* ln_b) {
  __shared__ __align__(16) __bf16 hl[16][HID];     // 16 KB
  __shared__ __align__(16) float yl[16][DD];       // 16 KB
#if HAVE_TDM
  __shared__ __align__(16) __bf16 wbuf[2 * HID * 32];  // 64 KB slice region
#endif
  int tile = blockIdx.x;                            // 0..255
  int wave = threadIdx.x >> 5, lane = threadIdx.x & 31;
  int half = lane >> 4, nl = lane & 15;
  const float* srow = S + (size_t)(tile * 16 + nl) * DD;
  const float* irow = incoming + (size_t)(tile * 16 + nl) * DD;

  // GEMM1: z=[S|incoming] (K=512) -> hidden 512; wave owns 128 hidden cols
  v8f acc[8];
#pragma unroll
  for (int t = 0; t < 8; ++t) acc[t] = vzero8();
#if HAVE_TDM
  unsigned wb0 = (unsigned)(uintptr_t)(const void*)&wbuf[0];
  unsigned wb1 = (unsigned)(uintptr_t)(const void*)&wbuf[HID * 32];
  if (wave == 0)                                   // w1 slice ks=0 -> buf0
    tdm_load_2d(wb0, w1, 32, HID, HID);
  for (int ks = 0; ks < 16; ++ks) {
    int kb = ks * 32;
    if (wave == 0) {
      if (ks + 1 < 16) {
        tdm_load_2d(((ks + 1) & 1) ? wb1 : wb0, w1 + (size_t)(kb + 32), 32,
                    HID, HID);
        __builtin_amdgcn_s_wait_tensorcnt(1);
      } else {
        __builtin_amdgcn_s_wait_tensorcnt(0);
      }
    }
    __syncthreads();
    const __bf16* ws = &wbuf[(ks & 1) * (HID * 32)];
    const float* zrow = (kb < DD) ? (srow + kb) : (irow + (kb - DD));
    bfrag a = make_a_frag_f32<false, false>(zrow, nullptr, 0, half);
    bfrag bc = load_b_frag(ws, 32, (wave * 8) * 16 + nl, 0, half);
#pragma unroll
    for (int t = 0; t < 8; ++t) {
      bfrag bn = (t < 7)
                     ? load_b_frag(ws, 32, (wave * 8 + t + 1) * 16 + nl, 0,
                                   half)
                     : bc;
      acc[t] = wmma_bf16(a, bc, acc[t]);
      bc = bn;
    }
    __syncthreads();
  }
  if (wave == 0)                                   // w2 slice ks=0 -> low buf
    tdm_load_2d(wb0, w2, 32, DD, HID);
#else
  for (int ks = 0; ks < 16; ++ks) {
    int kb = ks * 32;
    const float* zrow = (kb < DD) ? (srow + kb) : (irow + (kb - DD));
    bfrag a = make_a_frag_f32<false, false>(zrow, nullptr, 0, half);
    bfrag bc = load_b_frag(w1, HID, (wave * 8) * 16 + nl, kb, half);
#pragma unroll
    for (int t = 0; t < 8; ++t) {
      bfrag bn = (t < 7)
                     ? load_b_frag(w1, HID, (wave * 8 + t + 1) * 16 + nl, kb,
                                   half)
                     : bc;
      acc[t] = wmma_bf16(a, bc, acc[t]);
      bc = bn;
    }
  }
#endif
#pragma unroll
  for (int t = 0; t < 8; ++t) {
    int n = (wave * 8 + t) * 16 + nl;
    float bb = b1[n];
#pragma unroll
    for (int j = 0; j < 8; ++j)
      hl[j + 8 * half][n] = f2bf(geluf(acc[t][j] + bb));
  }
  __syncthreads();

  // GEMM2: hidden (K=512, bf16 in LDS) -> 256; wave owns 64 out cols
  v8f acc2[4];
#pragma unroll
  for (int t = 0; t < 4; ++t) acc2[t] = vzero8();
#if HAVE_TDM
  for (int ks = 0; ks < 16; ++ks) {
    if (wave == 0) {
      if (ks + 1 < 16) {
        tdm_load_2d(((ks + 1) & 1) ? (wb0 + DD * 32 * 2) : wb0,
                    w2 + (size_t)(ks * 32 + 32), 32, DD, HID);
        __builtin_amdgcn_s_wait_tensorcnt(1);
      } else {
        __builtin_amdgcn_s_wait_tensorcnt(0);
      }
    }
    __syncthreads();
    const __bf16* ws = &wbuf[(ks & 1) * (DD * 32)];
    int kb = ks * 32;
    bfrag a = load_a_frag_bf16(&hl[nl][0], kb, half);
    bfrag bc = load_b_frag(ws, 32, (wave * 4) * 16 + nl, 0, half);
#pragma unroll
    for (int t = 0; t < 4; ++t) {
      bfrag bn = (t < 3)
                     ? load_b_frag(ws, 32, (wave * 4 + t + 1) * 16 + nl, 0,
                                   half)
                     : bc;
      acc2[t] = wmma_bf16(a, bc, acc2[t]);
      bc = bn;
    }
    __syncthreads();
  }
#else
  for (int ks = 0; ks < 16; ++ks) {
    int kb = ks * 32;
    bfrag a = load_a_frag_bf16(&hl[nl][0], kb, half);
    bfrag bc = load_b_frag(w2, HID, (wave * 4) * 16 + nl, kb, half);
#pragma unroll
    for (int t = 0; t < 4; ++t) {
      bfrag bn = (t < 3)
                     ? load_b_frag(w2, HID, (wave * 4 + t + 1) * 16 + nl, kb,
                                   half)
                     : bc;
      acc2[t] = wmma_bf16(a, bc, acc2[t]);
      bc = bn;
    }
  }
#endif
#pragma unroll
  for (int t = 0; t < 4; ++t) {
    int n = (wave * 4 + t) * 16 + nl;
    float bb = b2[n];
#pragma unroll
    for (int j = 0; j < 8; ++j) {
      int mm = j + 8 * half;
      yl[mm][n] = S[(size_t)(tile * 16 + mm) * DD + n] + acc2[t][j] + bb;
    }
  }
  __syncthreads();

  // LayerNorm: 16 rows x 8 threads; intra-row reduce via shuffles (the 8
  // threads of a row group live in one wave under wave32).
  int rr = threadIdx.x >> 3, sub = threadIdx.x & 7;
  float s1 = 0.0f, s2 = 0.0f;
  for (int d = sub * 32; d < sub * 32 + 32; ++d) {
    float v = yl[rr][d];
    s1 += v; s2 += v * v;
  }
#pragma unroll
  for (int off = 1; off < 8; off <<= 1) {
    s1 += __shfl_xor(s1, off, 32);
    s2 += __shfl_xor(s2, off, 32);
  }
  float mean = s1 * (1.0f / 256.0f);
  float var = s2 * (1.0f / 256.0f) - mean * mean;
  float rstd = rsqrtf(var + 1e-5f);
  int grow = tile * 16 + rr;
  for (int d = sub * 32; d < sub * 32 + 32; ++d) {
    float v = (yl[rr][d] - mean) * rstd * ln_g[d] + ln_b[d];
    S[(size_t)grow * DD + d] = v;
  }
}

// --------------------------------- head ------------------------------------
__global__ __launch_bounds__(256) void k_head(const float* S, const int* qsrc,
                                              const int* qtgt, const float* w1,
                                              const float* b1, const float* w2,
                                              const float* b2, float* out) {
  __shared__ float in[2 * DD];
  __shared__ float h[DD];
  int b = blockIdx.x, t = threadIdx.x;
  int qs = min(max(qsrc[b], 0), NSLOT - 1);
  int qt = min(max(qtgt[b], 0), NSLOT - 1);
  in[t] = S[((size_t)b * NSLOT + qs) * DD + t];
  in[DD + t] = S[((size_t)b * NSLOT + qt) * DD + t];
  __syncthreads();
  float a = b1[t];
  const float* wr = w1 + t * (2 * DD);
  for (int k = 0; k < 2 * DD; ++k) a += wr[k] * in[k];
  h[t] = geluf(a);
  __syncthreads();
  if (t < 2) {
    float o = b2[t];
    const float* w2r = w2 + t * DD;
    for (int k = 0; k < DD; ++k) o += w2r[k] * h[k];
    out[b * 2 + t] = o;
  }
}

// ------------------------------ host driver --------------------------------
extern "C" void kernel_launch(void* const* d_in, const int* in_sizes, int n_in,
                              void* d_out, int out_size, void* d_ws,
                              size_t ws_size, hipStream_t stream) {
  (void)in_sizes; (void)n_in; (void)out_size; (void)ws_size;
  const int* src = (const int*)d_in[0];
  const int* rel = (const int*)d_in[1];
  const int* tgt = (const int*)d_in[2];
  // d_in[3] = mask (unused by reference)
  const int* qsrc = (const int*)d_in[4];
  const int* qtgt = (const int*)d_in[5];
  const float* node_embed = (const float*)d_in[6];
  const float* rel_embed  = (const float*)d_in[7];
  const float* edge_w1 = (const float*)d_in[8];
  const float* edge_b1 = (const float*)d_in[9];
  const float* edge_w2 = (const float*)d_in[10];
  const float* edge_b2 = (const float*)d_in[11];
  const float* query_w1 = (const float*)d_in[12];
  const float* query_b1 = (const float*)d_in[13];
  const float* query_w2 = (const float*)d_in[14];
  const float* query_b2 = (const float*)d_in[15];
  const float* group_w = (const float*)d_in[16];
  const float* slot_w  = (const float*)d_in[17];
  const float* msg_w1 = (const float*)d_in[18];
  const float* msg_b1 = (const float*)d_in[19];
  const float* msg_w2 = (const float*)d_in[20];
  const float* msg_b2 = (const float*)d_in[21];
  const float* gate_w1 = (const float*)d_in[22];
  const float* gate_b1 = (const float*)d_in[23];
  const float* gate_w2 = (const float*)d_in[24];
  const float* gate_b2 = (const float*)d_in[25];
  const float* upd_w1 = (const float*)d_in[26];
  const float* upd_b1 = (const float*)d_in[27];
  const float* upd_w2 = (const float*)d_in[28];
  const float* upd_b2 = (const float*)d_in[29];
  const float* ln_g = (const float*)d_in[30];
  const float* ln_b = (const float*)d_in[31];
  const float* S_init = (const float*)d_in[32];
  const float* head_w1 = (const float*)d_in[33];
  const float* head_b1 = (const float*)d_in[34];
  const float* head_w2 = (const float*)d_in[35];
  const float* head_b2 = (const float*)d_in[36];

  char* ws = (char*)d_ws;
  size_t cur = 0;
  auto take = [&](size_t bytes) -> char* {
    cur = (cur + 255) & ~(size_t)255;
    char* p = ws + cur;
    cur += bytes;
    return p;
  };
  float* S        = (float*)take((size_t)BB * NSLOT * DD * 4);
  float* incoming = (float*)take((size_t)BB * NSLOT * DD * 4);
  float* Hmsg  = (float*)take((size_t)BB * LTOT * HID * 4);   // 32 MB
  float* Hgate = (float*)take((size_t)BB * LTOT * DD * 4);    // 16 MB
  float* Hgrp  = (float*)take((size_t)BB * LTOT * 16 * 4);
  float* Hslt  = (float*)take((size_t)BB * LTOT * 16 * 4);
  float* sv_msg  = (float*)take((size_t)BB * HID * 4);
  float* sv_gate = (float*)take((size_t)BB * DD * 4);
  float* sv_grp  = (float*)take((size_t)BB * 16 * 4);
  float* sv_slt  = (float*)take((size_t)BB * 16 * 4);
  float* gatev = (float*)take((size_t)BB * LTOT * 4);
  int*   nidx  = (int*)take((size_t)BB * LTOT * 4);
  __bf16* Xb       = (__bf16*)take((size_t)BB * LTOT * DD * 2);
  __bf16* w_msg2   = (__bf16*)take((size_t)256 * 512 * 2);
  __bf16* w_msg1x  = (__bf16*)take((size_t)512 * 256 * 2);
  __bf16* w_gate1x = (__bf16*)take((size_t)256 * 256 * 2);
  __bf16* w_upd1   = (__bf16*)take((size_t)512 * 512 * 2);
  __bf16* w_upd2   = (__bf16*)take((size_t)256 * 512 * 2);
  __bf16* w_edge1  = (__bf16*)take((size_t)256 * 768 * 2);
  __bf16* w_edge2  = (__bf16*)take((size_t)256 * 256 * 2);
  __bf16* w_grpx   = (__bf16*)take((size_t)16 * 256 * 2);
  __bf16* w_sltx   = (__bf16*)take((size_t)16 * 256 * 2);

  k_cvt<<<256, 256, 0, stream>>>(w_msg2, msg_w2, 256 * 512);
  k_cvt_cols<<<256, 256, 0, stream>>>(w_msg1x, msg_w1, 512, 512, 0, 256);
  k_cvt_cols<<<256, 256, 0, stream>>>(w_gate1x, gate_w1, 256, 512, 0, 256);
  k_cvt_cols<<<16, 256, 0, stream>>>(w_grpx, group_w, 16, 512, 0, 256);
  k_cvt_cols<<<16, 256, 0, stream>>>(w_sltx, slot_w, 16, 512, 0, 256);
  k_cvt<<<256, 256, 0, stream>>>(w_upd1, upd_w1, 512 * 512);
  k_cvt<<<256, 256, 0, stream>>>(w_upd2, upd_w2, 256 * 512);
  k_cvt<<<256, 256, 0, stream>>>(w_edge1, edge_w1, 256 * 768);
  k_cvt<<<256, 256, 0, stream>>>(w_edge2, edge_w2, 256 * 256);

  k_init_S<<<1024, 256, 0, stream>>>(S, S_init);
  k_query<<<BB, 256, 0, stream>>>(qsrc, qtgt, node_embed, query_w1, query_b1,
                                  query_w2, query_b2, Xb);
  k_edge<<<256, 128, 0, stream>>>(src, rel, tgt, node_embed, rel_embed,
                                  w_edge1, edge_b1, w_edge2, edge_b2, Xb);

  const int M = BB * LTOT;  // 16384
  k_lin<<<8192, 128, 0, stream>>>(Xb, w_msg1x, msg_b1, Hmsg, M, 512, 256);
  k_lin<<<4096, 128, 0, stream>>>(Xb, w_gate1x, gate_b1, Hgate, M, 256, 256);
  k_lin<<<256, 128, 0, stream>>>(Xb, w_grpx, nullptr, Hgrp, M, 16, 256);
  k_lin<<<256, 128, 0, stream>>>(Xb, w_sltx, nullptr, Hslt, M, 16, 256);

  for (int it = 0; it < NSTEPS; ++it) {
    k_sv<<<BB, 256, 0, stream>>>(S, msg_w1, gate_w1, group_w, slot_w, sv_msg,
                                 sv_gate, sv_grp, sv_slt);
    k_zero<<<512, 256, 0, stream>>>(incoming, BB * NSLOT * DD);
    k_route<<<2048, 256, 0, stream>>>(Hgate, sv_gate, gate_w2, gate_b2, Hgrp,
                                      sv_grp, Hslt, sv_slt, gatev, nidx);
    k_msg<<<256, 128, 0, stream>>>(Hmsg, sv_msg, w_msg2, msg_b2, gatev, nidx,
                                   incoming);
    k_update<<<256, 128, 0, stream>>>(S, incoming, w_upd1, upd_b1, w_upd2,
                                      upd_b2, ln_g, ln_b);
  }
  k_head<<<BB, 256, 0, stream>>>(S, qsrc, qtgt, head_w1, head_b1, head_w2,
                                 head_b2, (float*)d_out);
}
